// Renet_31078383354533
// MI455X (gfx1250) — matrix operand order repaired and checked by reading, not compiled, not run on hardware
//
#include <hip/hip_runtime.h>

// ---------------------------------------------------------------------------
// ReNet (BiLSTM-V -> BiLSTM-H -> 1x1 conv -> BN) for MI455X / gfx1250.
// All GEMM-shaped math on v_wmma_f32_16x16x32_bf16; f32 cell state & gates.
// Gate nonlinearities use native v_exp_f32/v_rcp_f32 (branch-free) since they
// sit on the 96-step serial recurrence chain.
// ---------------------------------------------------------------------------

#define Bn   32
#define Cc   64
#define Ww   96
#define Hh   96
#define Ll   32
#define OUTC 256
#define NSEQ 3072       // B*W == B*H
#define Tt   96
#define Dd   64         // LSTM input width (C == 2L)
#define Gg   128        // 4L gate width

typedef __attribute__((ext_vector_type(16))) __bf16 v16bf;
typedef __attribute__((ext_vector_type(8)))  float  v8f;

union V16 { v16bf v; uint4 q[2]; };

__device__ __forceinline__ unsigned short f32_to_bf16(float f) {
    unsigned u = __float_as_uint(f);
    u += 0x7FFFu + ((u >> 16) & 1u);            // round-to-nearest-even
    return (unsigned short)(u >> 16);
}

__device__ __forceinline__ v8f wmma_bf16(const V16& a, const V16& b, v8f c) {
    return __builtin_amdgcn_wmma_f32_16x16x32_bf16(false, a.v, false, b.v,
                                                   (short)0, c, false, false);
}

#define LOG2E 1.4426950408889634f

// branch-free, native-HW sigmoid/tanh (v_exp_f32 + v_rcp_f32); correct
// saturation at +/-inf via exp2 overflow/underflow.
__device__ __forceinline__ float fast_sigmoid(float x) {
    return __builtin_amdgcn_rcpf(1.0f + __builtin_amdgcn_exp2f(-LOG2E * x));
}
__device__ __forceinline__ float fast_tanh(float x) {
    return 1.0f - 2.0f * __builtin_amdgcn_rcpf(
        1.0f + __builtin_amdgcn_exp2f((2.0f * LOG2E) * x));
}

// ---------------------------------------------------------------------------
// Pack x[B,C,W,H] f32 -> xv[(b*96+w)*96+h][64] bf16 (LDS tile transpose)
// grid (96, 6, 32), block 256
// ---------------------------------------------------------------------------
__global__ __launch_bounds__(256) void pack_x_kernel(
    const float* __restrict__ x, unsigned short* __restrict__ xv)
{
    const int w = blockIdx.x, b = blockIdx.z;
    const int h0 = blockIdx.y * 16;
    __shared__ __align__(16) unsigned short tile[16][72];
    #pragma unroll
    for (int it = 0; it < 4; ++it) {
        int c = it * 16 + (threadIdx.x >> 4);
        int h = threadIdx.x & 15;
        float v = x[(((size_t)(b * Cc + c) * Ww) + w) * Hh + h0 + h];
        tile[h][c] = f32_to_bf16(v);
    }
    __syncthreads();
    #pragma unroll
    for (int it = 0; it < 4; ++it) {
        int h = it * 4 + (threadIdx.x >> 6);
        int c = threadIdx.x & 63;
        xv[(((size_t)(b * Ww + w) * Hh) + h0 + h) * Dd + c] = tile[h][c];
    }
}

// f32 -> bf16 flat convert
__global__ void cvt_bf16_kernel(const float* __restrict__ in,
                                unsigned short* __restrict__ out, int n)
{
    int i = blockIdx.x * blockDim.x + threadIdx.x;
    if (i < n) out[i] = f32_to_bf16(in[i]);
}

// bias = b_ih + b_hh for fwd (0..127) and bwd (128..255); block 256
__global__ void add_bias_kernel(const float* __restrict__ fi, const float* __restrict__ fh,
                                const float* __restrict__ bi, const float* __restrict__ bh,
                                float* __restrict__ out)
{
    int i = threadIdx.x;
    out[i] = (i < Gg) ? (fi[i] + fh[i]) : (bi[i - Gg] + bh[i - Gg]);
}

// ---------------------------------------------------------------------------
// Bidirectional LSTM pass (one direction per blockIdx.y).
//   xin  : [seq][96][64] bf16     (seq = hi*96 + lo)
//   wih  : [2][128][64] bf16, whh: [2][128][32] bf16, bias: [2][128] f32
//   hout : bf16, element ((hi*9216 + tphys*t_stride + lo*lo_stride)*64 + j + 32*dir)
// vertical: (t_stride,lo_stride)=(96,1); horizontal: (1,96)
// One wave = 16 sequences x 128 gates; 3 WMMAs per N-tile per step.
// grid (48, 2), block 128 (4 waves)
// ---------------------------------------------------------------------------
__global__ __launch_bounds__(128) void lstm_pass_kernel(
    const unsigned short* __restrict__ xin,
    const unsigned short* __restrict__ wih,
    const unsigned short* __restrict__ whh,
    const float* __restrict__ bias,
    unsigned short* __restrict__ hout,
    int t_stride, int lo_stride)
{
    const int dir = blockIdx.y;
    __shared__ __align__(16) unsigned short s_wih[Gg * Dd];   // 16 KB
    __shared__ __align__(16) unsigned short s_whh[Gg * Ll];   //  8 KB
    __shared__ __align__(16) unsigned short s_h[4][16 * Ll];  //  4 KB

    const unsigned short* wih_d = wih + (size_t)dir * Gg * Dd;
    const unsigned short* whh_d = whh + (size_t)dir * Gg * Ll;
    for (int i = threadIdx.x; i < Gg * Dd / 8; i += blockDim.x)
        ((uint4*)s_wih)[i] = ((const uint4*)wih_d)[i];
    for (int i = threadIdx.x; i < Gg * Ll / 8; i += blockDim.x)
        ((uint4*)s_whh)[i] = ((const uint4*)whh_d)[i];
    __syncthreads();

    const int wave = threadIdx.x >> 5;
    const int lane = threadIdx.x & 31;
    const int half = lane >> 4;     // 0: M=r / K 0..7,16..23 ; 1: M=8+r / +8
    const int n    = lane & 15;     // N column within a tile
    const int seqBase = (blockIdx.x * 4 + wave) * 16;

    unsigned short* hl = s_h[wave];
    for (int i = lane; i < 16 * Ll; i += 32) hl[i] = 0;   // h0 = 0

    // bias broadcast per N-tile (depends only on gate column)
    float bias_v[8];
    #pragma unroll
    for (int tn = 0; tn < 8; ++tn) bias_v[tn] = bias[dir * Gg + tn * 16 + n];

    // whh B operands resident in VGPRs (K=32 -> one WMMA per tile)
    V16 bwh[8];
    #pragma unroll
    for (int tn = 0; tn < 8; ++tn) {
        const unsigned short* p = s_whh + (tn * 16 + n) * Ll + 8 * half;
        bwh[tn].q[0] = *(const uint4*)(p);
        bwh[tn].q[1] = *(const uint4*)(p + 16);
    }

    float cst[16];                       // cell state: [tp*8 + r]
    #pragma unroll
    for (int i = 0; i < 16; ++i) cst[i] = 0.0f;

    // per-reg output row bases (t-independent part)
    int rowc[8];
    #pragma unroll
    for (int r = 0; r < 8; ++r) {
        int seq = seqBase + r + 8 * half;
        rowc[r] = (seq / 96) * 9216 + (seq % 96) * lo_stride;
    }
    const int mrow = n;            // A-operand row supplied by this lane
    const int kofs = half * 8;     // A-operand K offset
    const int j0   = n + 32 * dir; // output feature base

    for (int t = 0; t < Tt; ++t) {
        const int tphys = dir ? (Tt - 1 - t) : t;
        // ---- A_x : 16x64 bf16 tile of the input sequence at time tphys
        const unsigned short* xr =
            xin + ((size_t)(seqBase + mrow) * Tt + tphys) * Dd + kofs;
        V16 ax0, ax1;
        ax0.q[0] = *(const uint4*)(xr);
        ax0.q[1] = *(const uint4*)(xr + 16);
        ax1.q[0] = *(const uint4*)(xr + 32);
        ax1.q[1] = *(const uint4*)(xr + 48);
        if (t < Tt - 1) {
            const int tnx = dir ? (tphys - 1) : (tphys + 1);
            __builtin_prefetch(xin + ((size_t)(seqBase + mrow) * Tt + tnx) * Dd, 0, 0);
        }
        // ---- A_h : 16x32 bf16 tile of h(t-1) from wave-private LDS
        const unsigned short* hr = hl + mrow * Ll + kofs;
        V16 ah;
        ah.q[0] = *(const uint4*)(hr);
        ah.q[1] = *(const uint4*)(hr + 16);

        // ---- gates = bias + x*wih^T + h*whh^T   (8 N-tiles x 3 WMMAs)
        v8f acc[8];
        #pragma unroll
        for (int tn = 0; tn < 8; ++tn) {
            v8f a;
            #pragma unroll
            for (int i = 0; i < 8; ++i) a[i] = bias_v[tn];
            const unsigned short* wp = s_wih + (tn * 16 + n) * Dd + 8 * half;
            V16 b0, b1;
            b0.q[0] = *(const uint4*)(wp);
            b0.q[1] = *(const uint4*)(wp + 16);
            b1.q[0] = *(const uint4*)(wp + 32);
            b1.q[1] = *(const uint4*)(wp + 48);
            a = wmma_bf16(ax0, b0, a);
            a = wmma_bf16(ax1, b1, a);
            a = wmma_bf16(ah, bwh[tn], a);
            acc[tn] = a;
        }

        // ---- elementwise LSTM update; i:tiles 0-1, f:2-3, g:4-5, o:6-7
        #pragma unroll
        for (int tp = 0; tp < 2; ++tp) {
            #pragma unroll
            for (int r = 0; r < 8; ++r) {
                float ig = fast_sigmoid(acc[0 + tp][r]);
                float fg = fast_sigmoid(acc[2 + tp][r]);
                float gg = fast_tanh(acc[4 + tp][r]);
                float og = fast_sigmoid(acc[6 + tp][r]);
                float c  = fg * cst[tp * 8 + r] + ig * gg;
                cst[tp * 8 + r] = c;
                float hv = og * fast_tanh(c);
                unsigned short hb = f32_to_bf16(hv);
                int m = r + 8 * half;
                hl[m * Ll + tp * 16 + n] = hb;                 // next step's A_h
                size_t row = (size_t)(rowc[r] + tphys * t_stride);
                hout[row * Dd + j0 + tp * 16] = hb;            // global output
            }
        }
    }
}

// ---------------------------------------------------------------------------
// 1x1 conv GEMM: y[b,o,w,h] = sum_j hh[(b*96+h)*96+w][j] * cw[o][j] + cb[o]
// one wave = 16 consecutive h x 256 outputs (16 N-tiles); grid 4608, block 128
// ---------------------------------------------------------------------------
__global__ __launch_bounds__(128) void conv1x1_kernel(
    const unsigned short* __restrict__ hin,  // [(b*96+h)*96+w][64] bf16
    const unsigned short* __restrict__ cw,   // [256][64] bf16
    const float* __restrict__ cb,            // [256]
    float* __restrict__ y)                   // [B][256][96][96]
{
    const int wave = threadIdx.x >> 5;
    const int lane = threadIdx.x & 31;
    const int half = lane >> 4;
    const int n    = lane & 15;
    const int tile = blockIdx.x * 4 + wave;          // 32*96*6 tiles
    const int b    = tile / (Ww * 6);
    const int rem  = tile % (Ww * 6);
    const int w    = rem / 6;
    const int h0   = (rem % 6) * 16;

    const unsigned short* xr =
        hin + ((size_t)((b * Hh + h0 + n) * Ww) + w) * Dd + half * 8;
    V16 ax0, ax1;
    ax0.q[0] = *(const uint4*)(xr);
    ax0.q[1] = *(const uint4*)(xr + 16);
    ax1.q[0] = *(const uint4*)(xr + 32);
    ax1.q[1] = *(const uint4*)(xr + 48);

    #pragma unroll 4
    for (int tn = 0; tn < 16; ++tn) {
        const int o = tn * 16 + n;
        v8f a;
        float bv = cb[o];
        #pragma unroll
        for (int i = 0; i < 8; ++i) a[i] = bv;
        const unsigned short* wp = cw + (size_t)o * Dd + 8 * half;
        V16 b0, b1;
        b0.q[0] = *(const uint4*)(wp);
        b0.q[1] = *(const uint4*)(wp + 16);
        b1.q[0] = *(const uint4*)(wp + 32);
        b1.q[1] = *(const uint4*)(wp + 48);
        a = wmma_bf16(ax0, b0, a);
        a = wmma_bf16(ax1, b1, a);
        float* op = y + (((size_t)(b * OUTC + o) * Ww + w) * Hh + h0 + 8 * half);
        *(float4*)(op)     = make_float4(a[0], a[1], a[2], a[3]);
        *(float4*)(op + 4) = make_float4(a[4], a[5], a[6], a[7]);
    }
}

// ---------------------------------------------------------------------------
// BN batch stats per channel -> scale/shift; grid 256, block 256
// ---------------------------------------------------------------------------
__global__ __launch_bounds__(256) void bn_stats_kernel(
    const float* __restrict__ y, const float* __restrict__ g,
    const float* __restrict__ bb, float* __restrict__ scale,
    float* __restrict__ shift)
{
    const int o = blockIdx.x;
    float s = 0.0f, s2 = 0.0f;
    for (int b = 0; b < Bn; ++b) {
        const float* p = y + (size_t)(b * OUTC + o) * (Ww * Hh);
        for (int i = threadIdx.x; i < Ww * Hh; i += 256) {
            float v = p[i];
            s += v; s2 += v * v;
        }
    }
    __shared__ float rs[256], rs2[256];
    rs[threadIdx.x] = s; rs2[threadIdx.x] = s2;
    __syncthreads();
    for (int st = 128; st > 0; st >>= 1) {
        if (threadIdx.x < st) {
            rs[threadIdx.x]  += rs[threadIdx.x + st];
            rs2[threadIdx.x] += rs2[threadIdx.x + st];
        }
        __syncthreads();
    }
    if (threadIdx.x == 0) {
        const float inv_n = 1.0f / (float)(Bn * Ww * Hh);
        float mean = rs[0] * inv_n;
        float var  = rs2[0] * inv_n - mean * mean;
        float sc   = g[o] * rsqrtf(var + 1e-5f);
        scale[o] = sc;
        shift[o] = bb[o] - mean * sc;
    }
}

// in-place normalize; float4 over y
__global__ void bn_apply_kernel(float* __restrict__ y,
                                const float* __restrict__ scale,
                                const float* __restrict__ shift, size_t n4)
{
    size_t i = (size_t)blockIdx.x * blockDim.x + threadIdx.x;
    if (i >= n4) return;
    int o = (int)((i / (Ww * Hh / 4)) % OUTC);
    float sc = scale[o], sh = shift[o];
    float4 v = ((float4*)y)[i];
    v.x = v.x * sc + sh; v.y = v.y * sc + sh;
    v.z = v.z * sc + sh; v.w = v.w * sc + sh;
    ((float4*)y)[i] = v;
}

// ---------------------------------------------------------------------------
extern "C" void kernel_launch(void* const* d_in, const int* in_sizes, int n_in,
                              void* d_out, int out_size, void* d_ws, size_t ws_size,
                              hipStream_t stream)
{
    const float* x       = (const float*)d_in[0];
    const float* v_wih_f = (const float*)d_in[1];
    const float* v_whh_f = (const float*)d_in[2];
    const float* v_bih_f = (const float*)d_in[3];
    const float* v_bhh_f = (const float*)d_in[4];
    const float* v_wih_b = (const float*)d_in[5];
    const float* v_whh_b = (const float*)d_in[6];
    const float* v_bih_b = (const float*)d_in[7];
    const float* v_bhh_b = (const float*)d_in[8];
    const float* h_wih_f = (const float*)d_in[9];
    const float* h_whh_f = (const float*)d_in[10];
    const float* h_bih_f = (const float*)d_in[11];
    const float* h_bhh_f = (const float*)d_in[12];
    const float* h_wih_b = (const float*)d_in[13];
    const float* h_whh_b = (const float*)d_in[14];
    const float* h_bih_b = (const float*)d_in[15];
    const float* h_bhh_b = (const float*)d_in[16];
    const float* conv_w  = (const float*)d_in[17];
    const float* conv_b  = (const float*)d_in[18];
    const float* bn_g    = (const float*)d_in[19];
    const float* bn_b    = (const float*)d_in[20];
    float* out = (float*)d_out;

    // workspace carve-up
    char* w = (char*)d_ws;
    const size_t SEQBYTES = (size_t)NSEQ * Tt * Dd * sizeof(unsigned short); // 37.75 MB
    unsigned short* xv  = (unsigned short*)(w);
    unsigned short* vbf = (unsigned short*)(w + SEQBYTES);
    unsigned short* hhb = (unsigned short*)(w + 2 * SEQBYTES);
    size_t o = 3 * SEQBYTES;
    unsigned short* w_vih = (unsigned short*)(w + o); o += 2 * Gg * Dd * 2;
    unsigned short* w_vhh = (unsigned short*)(w + o); o += 2 * Gg * Ll * 2;
    unsigned short* w_hih = (unsigned short*)(w + o); o += 2 * Gg * Dd * 2;
    unsigned short* w_hhh = (unsigned short*)(w + o); o += 2 * Gg * Ll * 2;
    unsigned short* w_cw  = (unsigned short*)(w + o); o += OUTC * Dd * 2;
    float* b_v   = (float*)(w + o); o += 1024;
    float* b_h   = (float*)(w + o); o += 1024;
    float* scale = (float*)(w + o); o += 1024;
    float* shift = (float*)(w + o); o += 1024;

    // 1) pack x -> bf16 vertical-sequence layout
    pack_x_kernel<<<dim3(Ww, Hh / 16, Bn), 256, 0, stream>>>(x, xv);

    // 2) pack weights to bf16, fold biases
    cvt_bf16_kernel<<<32, 256, 0, stream>>>(v_wih_f, w_vih,            Gg * Dd);
    cvt_bf16_kernel<<<32, 256, 0, stream>>>(v_wih_b, w_vih + Gg * Dd,  Gg * Dd);
    cvt_bf16_kernel<<<16, 256, 0, stream>>>(v_whh_f, w_vhh,            Gg * Ll);
    cvt_bf16_kernel<<<16, 256, 0, stream>>>(v_whh_b, w_vhh + Gg * Ll,  Gg * Ll);
    cvt_bf16_kernel<<<32, 256, 0, stream>>>(h_wih_f, w_hih,            Gg * Dd);
    cvt_bf16_kernel<<<32, 256, 0, stream>>>(h_wih_b, w_hih + Gg * Dd,  Gg * Dd);
    cvt_bf16_kernel<<<16, 256, 0, stream>>>(h_whh_f, w_hhh,            Gg * Ll);
    cvt_bf16_kernel<<<16, 256, 0, stream>>>(h_whh_b, w_hhh + Gg * Ll,  Gg * Ll);
    cvt_bf16_kernel<<<64, 256, 0, stream>>>(conv_w,  w_cw,             OUTC * Dd);
    add_bias_kernel<<<1, 256, 0, stream>>>(v_bih_f, v_bhh_f, v_bih_b, v_bhh_b, b_v);
    add_bias_kernel<<<1, 256, 0, stream>>>(h_bih_f, h_bhh_f, h_bih_b, h_bhh_b, b_h);

    // 3) vertical BiLSTM (seq=b*96+w, t=h): out row = b*9216 + h*96 + w
    lstm_pass_kernel<<<dim3(48, 2), 128, 0, stream>>>(
        xv, w_vih, w_vhh, b_v, vbf, /*t_stride=*/96, /*lo_stride=*/1);

    // 4) horizontal BiLSTM (seq=b*96+h, t=w): out row = b*9216 + h*96 + w
    lstm_pass_kernel<<<dim3(48, 2), 128, 0, stream>>>(
        vbf, w_hih, w_hhh, b_h, hhb, /*t_stride=*/1, /*lo_stride=*/96);

    // 5) 1x1 conv GEMM -> y in d_out
    conv1x1_kernel<<<(Bn * Ww * 6) / 4, 128, 0, stream>>>(hhb, w_cw, conv_b, out);

    // 6) BN stats + in-place normalize
    bn_stats_kernel<<<OUTC, 256, 0, stream>>>(out, bn_g, bn_b, scale, shift);
    const size_t n4 = (size_t)Bn * OUTC * Ww * Hh / 4;
    bn_apply_kernel<<<(unsigned)((n4 + 255) / 256), 256, 0, stream>>>(out, scale, shift, n4);
}
